// CasualAttentionLayer_30064771072893
// MI455X (gfx1250) — matrix operand order repaired and checked by reading, not compiled, not run on hardware
//
#include <hip/hip_runtime.h>
#include <stdint.h>

typedef _Float16 half_t;
typedef __attribute__((ext_vector_type(16))) _Float16 v16h;
typedef __attribute__((ext_vector_type(8)))  _Float16 v8h;
typedef __attribute__((ext_vector_type(8)))  float    v8f;

#define B_DIM 4
#define S_DIM 2048
#define D_DIM 1024

static __device__ __forceinline__ v16h cat16(v8h lo, v8h hi) {
  v16h r;
#pragma unroll
  for (int j = 0; j < 8; ++j) { r[j] = lo[j]; r[j + 8] = hi[j]; }
  return r;
}

static __device__ __forceinline__ v8f wmma_f16(v16h a, v16h b, v8f c) {
  // D = A(16x32) * B(32x16) + C, f32 accumulate
  return __builtin_amdgcn_wmma_f32_16x16x32_f16(false, a, false, b, (short)0, c,
                                                false, false);
}

// Async copy of 32B (two B128 ops) global -> LDS, tracked by ASYNCcnt.
static __device__ __forceinline__ void async_copy_32B(uint32_t lds_addr,
                                                      const void* gaddr) {
  asm volatile("global_load_async_to_lds_b128 %0, %1, off"
               :: "v"(lds_addr), "v"(gaddr) : "memory");
  asm volatile("global_load_async_to_lds_b128 %0, %1, off offset:16"
               :: "v"(lds_addr), "v"(gaddr) : "memory");
}

// ---------------------------------------------------------------------------
// Kernel 0a: X f32 -> f16, elementwise.  8M elements, 8 per thread.
// ---------------------------------------------------------------------------
__global__ __launch_bounds__(256) void cvt_x_kernel(
    const float* __restrict__ X, half_t* __restrict__ X16)
{
  const size_t i = ((size_t)blockIdx.x * 256 + threadIdx.x) * 8;
  const float4* src = (const float4*)(X + i);
  const float4 a = src[0];
  const float4 b = src[1];
  v8h o;
  o[0] = (half_t)a.x; o[1] = (half_t)a.y; o[2] = (half_t)a.z; o[3] = (half_t)a.w;
  o[4] = (half_t)b.x; o[5] = (half_t)b.y; o[6] = (half_t)b.z; o[7] = (half_t)b.w;
  *(v8h*)(X16 + i) = o;
}

// ---------------------------------------------------------------------------
// Kernel 0b: W [in,out] f32 -> WT [out,in] f16, LDS-tiled 32x32 transpose.
// grid = (1024/32, 1024/32, 3); block = 256.
// ---------------------------------------------------------------------------
__global__ __launch_bounds__(256) void cvt_w_kernel(
    const float* __restrict__ Wq, const float* __restrict__ Wk,
    const float* __restrict__ Wv,
    half_t* __restrict__ WTq, half_t* __restrict__ WTk, half_t* __restrict__ WTv)
{
  __shared__ half_t tile[32][33];   // +1 pad to dodge bank conflicts
  const int z = blockIdx.z;
  const float* W  = (z == 0) ? Wq  : (z == 1) ? Wk  : Wv;
  half_t*      WT = (z == 0) ? WTq : (z == 1) ? WTk : WTv;

  const int k0 = blockIdx.x * 32;   // W row block   (input dim)
  const int n0 = blockIdx.y * 32;   // W col block   (output dim)
  const int c  = threadIdx.x & 31;
  const int r0 = threadIdx.x >> 5;  // 0..7

#pragma unroll
  for (int rr = 0; rr < 32; rr += 8)
    tile[r0 + rr][c] = (half_t)W[(size_t)(k0 + r0 + rr) * D_DIM + n0 + c];
  __syncthreads();
#pragma unroll
  for (int rr = 0; rr < 32; rr += 8)
    WT[(size_t)(n0 + r0 + rr) * D_DIM + k0 + c] = tile[c][r0 + rr];
}

// ---------------------------------------------------------------------------
// Kernel 1: QKV projection, all-f16 operands.  out = X @ W + b.
// grid = (8192/128, 1024/128, 3); block = 256 (8 waves).
// Wave w: rows [bx*128 + 16w, +16), cols [by*128, +128) = 8 accum tiles.
// The 128x32 B-tile (shared by all 8 waves) is double-buffered in LDS via
// async global->LDS copies (ASYNCcnt); B-frags then come from ds_load_b128,
// software-pipelined one fragment ahead of the consuming WMMA.
// z==2 stores V transposed [B,D,S] so attention's B-operand is contiguous.
// ---------------------------------------------------------------------------
#define WROW 40   // LDS row stride in halfs (80B): 16B-aligned, bank-spread

__global__ __launch_bounds__(256) void qkv_proj_kernel(
    const half_t* __restrict__ X16,
    const half_t* __restrict__ WTq, const float* __restrict__ bq,
    const half_t* __restrict__ WTk, const float* __restrict__ bk,
    const half_t* __restrict__ WTv, const float* __restrict__ bv,
    half_t* __restrict__ Qo, half_t* __restrict__ Ko, half_t* __restrict__ VTo)
{
  __shared__ alignas(16) half_t Wtile[2][128][WROW];

  const int tid  = threadIdx.x;
  const int lane = tid & 31;
  const int wave = tid >> 5;
  const int h    = lane >> 4;
  const int ln   = lane & 15;
  const int z    = blockIdx.z;

  const half_t* WT   = (z == 0) ? WTq : (z == 1) ? WTk : WTv;
  const float*  bias = (z == 0) ? bq  : (z == 1) ? bk  : bv;

  const int mrow = blockIdx.x * 128 + wave * 16;
  const int n0   = blockIdx.y * 128;

  const half_t* xrow = X16 + (size_t)(mrow + ln) * D_DIM;  // A row (M = ln)

  // this thread's slice of the async-staged B tile: row r, 16-half chunk cb
  const int r_st  = tid >> 1;
  const int cb_st = (tid & 1) * 16;
  const half_t* wsrc_base = WT + (size_t)(n0 + r_st) * D_DIM + cb_st;
  const uint32_t ldst0 = (uint32_t)(uintptr_t)&Wtile[0][r_st][cb_st];
  const uint32_t ldst1 = (uint32_t)(uintptr_t)&Wtile[1][r_st][cb_st];

  v8f z8 = {0.f, 0.f, 0.f, 0.f, 0.f, 0.f, 0.f, 0.f};
  v8f acc[8];
#pragma unroll
  for (int ni = 0; ni < 8; ++ni) acc[ni] = z8;

  // prologue: stage k-tile 0 into buffer 0
  async_copy_32B(ldst0, wsrc_base);

  int buf = 0;
  for (int k0 = 0; k0 < D_DIM; k0 += 32) {
    if (k0 + 32 < D_DIM) {
      // prefetch next k-tile into the other buffer
      async_copy_32B(buf ? ldst0 : ldst1, wsrc_base + k0 + 32);
      asm volatile("s_wait_asynccnt 0x2" ::: "memory");  // current buf landed
    } else {
      asm volatile("s_wait_asynccnt 0x0" ::: "memory");
    }
    __syncthreads();

    // A fragment: lane m=ln, k = k0 + 16h + j -> two 16B loads
    v8h alo = *(const v8h*)(xrow + k0 + 16 * h);
    v8h ahi = *(const v8h*)(xrow + k0 + 16 * h + 8);
    const v16h afrag = cat16(alo, ahi);

    // B fragment from LDS: lane n=ln, k = 16h + j -> two ds_load_b128.
    // One-deep rotation so load(ni+1) overlaps wmma(ni).
    auto loadB = [&](int ni) -> v16h {
      const half_t* brow = &Wtile[buf][ni * 16 + ln][16 * h];
      v8h blo = *(const v8h*)(brow);
      v8h bhi = *(const v8h*)(brow + 8);
      return cat16(blo, bhi);
    };
    v16h bcur = loadB(0);
#pragma unroll
    for (int ni = 0; ni < 8; ++ni) {
      v16h bnxt = (ni < 7) ? loadB(ni + 1) : bcur;
      acc[ni] = wmma_f16(afrag, bcur, acc[ni]);
      bcur = bnxt;
    }
    __syncthreads();   // all waves done reading buf before it is re-filled
    buf ^= 1;
  }

  // C/D layout: VGPR r, lanes 0-15 -> row r, lanes 16-31 -> row r+8, col = ln
#pragma unroll
  for (int ni = 0; ni < 8; ++ni) {
    const int col = n0 + ni * 16 + ln;
    const float bn = bias[col];
    if (z < 2) {
      half_t* O = (z == 0) ? Qo : Ko;
#pragma unroll
      for (int r = 0; r < 8; ++r) {
        const int row = mrow + 8 * h + r;
        O[(size_t)row * D_DIM + col] = (half_t)(acc[ni][r] + bn);
      }
    } else {
      v8h hv;
#pragma unroll
      for (int r = 0; r < 8; ++r) hv[r] = (half_t)(acc[ni][r] + bn);
      const int batch = mrow / S_DIM;
      const int srow  = (mrow % S_DIM) + 8 * h;
      *(v8h*)(VTo + (size_t)batch * D_DIM * S_DIM + (size_t)col * S_DIM + srow) = hv;
    }
  }
}

// ---------------------------------------------------------------------------
// Kernel 2: causal flash attention (no 1/sqrt(d) scaling, per reference).
// grid = (S/32, B); block = 256 (8 waves).  Block owns 32 query rows.
// Wave w owns D-slice [128w, 128w+128): Q frags + O accum live in registers.
// 64-key tiles: split-K scores -> ds_add_f32 into padded LDS tile, wave0
// online softmax (two-pass scan), O rescale, then P@V from transposed V.
// ---------------------------------------------------------------------------
#define SROW 68   // Stile row stride (floats): 16B-aligned, bank-rotated
#define PROW 72   // Ptile row stride (halfs):  16B-aligned, bank-rotated

__global__ __launch_bounds__(256) void flash_attn_kernel(
    const half_t* __restrict__ Q, const half_t* __restrict__ K,
    const half_t* __restrict__ VT, float* __restrict__ Out)
{
  __shared__ alignas(16) float  Stile[32][SROW];
  __shared__ alignas(16) half_t Ptile[32][PROW];
  __shared__ float rowmax[32];
  __shared__ float rowsum[32];
  __shared__ float corr[32];

  const int tid  = threadIdx.x;
  const int lane = tid & 31;
  const int wave = tid >> 5;
  const int h    = lane >> 4;
  const int ln   = lane & 15;

  const int qbase = blockIdx.x * 32;
  const int batch = blockIdx.y;
  const size_t bofs = (size_t)batch * S_DIM * D_DIM;

  // Persistent Q A-fragments over this wave's K-slice (k = 128*wave + 32*ks)
  v16h qf[2][4];
#pragma unroll
  for (int mi = 0; mi < 2; ++mi)
#pragma unroll
    for (int ks = 0; ks < 4; ++ks) {
      const half_t* qrow = Q + bofs +
          (size_t)(qbase + mi * 16 + ln) * D_DIM + wave * 128 + ks * 32;
      v8h lo = *(const v8h*)(qrow + 16 * h);
      v8h hi = *(const v8h*)(qrow + 16 * h + 8);
      qf[mi][ks] = cat16(lo, hi);
    }

  v8f z8 = {0.f, 0.f, 0.f, 0.f, 0.f, 0.f, 0.f, 0.f};
  v8f oacc[2][8];
#pragma unroll
  for (int mi = 0; mi < 2; ++mi)
#pragma unroll
    for (int ni = 0; ni < 8; ++ni) oacc[mi][ni] = z8;

  if (tid < 32) { rowmax[tid] = -__builtin_inff(); rowsum[tid] = 0.f; }
  __syncthreads();

  const int nkt = (qbase + 95) >> 6;  // 64-key tiles covering keys <= qbase+31
  for (int kt = 0; kt < nkt; ++kt) {
    const int key0 = kt * 64;

    // zero the score tile: 32*68 floats = 544 float4
    {
      float4 z4 = {0.f, 0.f, 0.f, 0.f};
      float4* sz = (float4*)&Stile[0][0];
      sz[tid] = z4;
      sz[tid + 256] = z4;
      if (tid < 32) sz[tid + 512] = z4;
    }
    __syncthreads();

    // split-K partial scores: this wave contributes K-range [128w, 128w+128)
#pragma unroll
    for (int ni = 0; ni < 4; ++ni) {
      v16h kf[4];
#pragma unroll
      for (int ks = 0; ks < 4; ++ks) {
        const half_t* krow = K + bofs +
            (size_t)(key0 + ni * 16 + ln) * D_DIM + wave * 128 + ks * 32;
        kf[ks] = *(const v16h*)(krow + 16 * h);  // B-frag: k = 16h + j contiguous
      }
#pragma unroll
      for (int mi = 0; mi < 2; ++mi) {
        v8f s = z8;
#pragma unroll
        for (int ks = 0; ks < 4; ++ks) s = wmma_f16(qf[mi][ks], kf[ks], s);
#pragma unroll
        for (int r = 0; r < 8; ++r)
          atomicAdd(&Stile[mi * 16 + 8 * h + r][ni * 16 + ln], s[r]);
      }
    }
    __syncthreads();

    // online softmax: wave 0, one lane per query row, two-pass scan
    if (tid < 32) {
      const int r = tid;
      const int q = qbase + r;
      const float m_old = rowmax[r];
      float mx = m_old;
#pragma unroll 8
      for (int c = 0; c < 64; ++c) {
        float s = Stile[r][c];
        if (key0 + c > q) s = -__builtin_inff();  // causal mask
        mx = fmaxf(mx, s);
      }
      float sum = 0.f;
#pragma unroll 8
      for (int c = 0; c < 64; ++c) {
        float s = Stile[r][c];
        if (key0 + c > q) s = -__builtin_inff();
        const float p = __expf(s - mx);           // masked -> 0
        sum += p;
        Ptile[r][c] = (half_t)p;
      }
      const float cr = __expf(m_old - mx);        // first tile: exp(-inf) = 0
      corr[r]   = cr;
      rowmax[r] = mx;
      rowsum[r] = rowsum[r] * cr + sum;
    }
    __syncthreads();

    // rescale O accumulators by per-row correction
    float cf[2][8];
#pragma unroll
    for (int mi = 0; mi < 2; ++mi)
#pragma unroll
      for (int r = 0; r < 8; ++r) cf[mi][r] = corr[mi * 16 + 8 * h + r];
#pragma unroll
    for (int mi = 0; mi < 2; ++mi)
#pragma unroll
      for (int ni = 0; ni < 8; ++ni)
#pragma unroll
        for (int r = 0; r < 8; ++r) oacc[mi][ni][r] *= cf[mi][r];

    // O += P(32x64) @ V(64 x 128-slice): two 32-deep K steps
#pragma unroll
    for (int kk = 0; kk < 2; ++kk) {
      v16h pf[2];
#pragma unroll
      for (int mi = 0; mi < 2; ++mi) {
        const half_t* prow = &Ptile[mi * 16 + ln][kk * 32];
        v8h lo = *(const v8h*)(prow + 16 * h);
        v8h hi = *(const v8h*)(prow + 16 * h + 8);
        pf[mi] = cat16(lo, hi);
      }
#pragma unroll
      for (int ni = 0; ni < 8; ++ni) {
        const half_t* vrow = VT + bofs +
            (size_t)(wave * 128 + ni * 16 + ln) * S_DIM + key0 + kk * 32;
        v16h bf = *(const v16h*)(vrow + 16 * h);  // contiguous: V^T layout
#pragma unroll
        for (int mi = 0; mi < 2; ++mi)
          oacc[mi][ni] = wmma_f16(pf[mi], bf, oacc[mi][ni]);
      }
    }
  }

  // normalize by row sums and write f32 output [B,S,D]
  float inv[2][8];
#pragma unroll
  for (int mi = 0; mi < 2; ++mi)
#pragma unroll
    for (int r = 0; r < 8; ++r) inv[mi][r] = 1.0f / rowsum[mi * 16 + 8 * h + r];

#pragma unroll
  for (int mi = 0; mi < 2; ++mi)
#pragma unroll
    for (int ni = 0; ni < 8; ++ni) {
      const int col = wave * 128 + ni * 16 + ln;
#pragma unroll
      for (int r = 0; r < 8; ++r) {
        const int row = qbase + mi * 16 + 8 * h + r;
        Out[bofs + (size_t)row * D_DIM + col] = oacc[mi][ni][r] * inv[mi][r];
      }
    }
}

// ---------------------------------------------------------------------------
extern "C" void kernel_launch(void* const* d_in, const int* in_sizes, int n_in,
                              void* d_out, int out_size, void* d_ws, size_t ws_size,
                              hipStream_t stream) {
  (void)in_sizes; (void)n_in; (void)out_size; (void)ws_size;
  const float* X  = (const float*)d_in[0];
  const float* Wq = (const float*)d_in[1];
  const float* bq = (const float*)d_in[2];
  const float* Wk = (const float*)d_in[3];
  const float* bk = (const float*)d_in[4];
  const float* Wv = (const float*)d_in[5];
  const float* bv = (const float*)d_in[6];
  float* Out = (float*)d_out;

  const size_t elems = (size_t)B_DIM * S_DIM * D_DIM;  // 8M
  const size_t welems = (size_t)D_DIM * D_DIM;         // 1M
  half_t* Qw  = (half_t*)d_ws;          // 16 MB
  half_t* Kw  = Qw + elems;             // 16 MB
  half_t* VTw = Kw + elems;             // 16 MB (V transposed [B,D,S])
  half_t* X16 = VTw + elems;            // 16 MB
  half_t* WTq = X16 + elems;            // 2 MB each
  half_t* WTk = WTq + welems;
  half_t* WTv = WTk + welems;

  cvt_x_kernel<<<dim3((unsigned)(elems / (256 * 8))), 256, 0, stream>>>(X, X16);

  dim3 gw(D_DIM / 32, D_DIM / 32, 3);
  cvt_w_kernel<<<gw, 256, 0, stream>>>(Wq, Wk, Wv, WTq, WTk, WTv);

  dim3 gqkv((B_DIM * S_DIM) / 128, D_DIM / 128, 3);
  qkv_proj_kernel<<<gqkv, 256, 0, stream>>>(X16, WTq, bq, WTk, bk, WTv, bv,
                                            Qw, Kw, VTw);

  dim3 gattn(S_DIM / 32, B_DIM);
  flash_attn_kernel<<<gattn, 256, 0, stream>>>(Qw, Kw, VTw, Out);
}
